// GCN_13846974562747
// MI455X (gfx1250) — matrix operand order repaired and checked by reading, not compiled, not run on hardware
//
#include <hip/hip_runtime.h>
#include <stdint.h>

typedef __attribute__((ext_vector_type(16))) _Float16 v16h;
typedef __attribute__((ext_vector_type(8)))  float    v8f;

#define KDIM 128   // both layers contract over 128

union AFrag { v16h v; uint32_t u[8]; };
union CFrag { v8f  v; float    f[8]; };
union PackU { _Float16 h[2]; uint32_t u; };

// ---------------- utility: zero a float range (grid-stride) ----------------
__global__ void k_zero(float* __restrict__ p, long n) {
  long i = (long)blockIdx.x * blockDim.x + threadIdx.x;
  long stride = (long)gridDim.x * blockDim.x;
  for (; i < n; i += stride) p[i] = 0.0f;
}

// ---------------- pack f32 weights [K][N] -> f16 K-pair dwords -------------
// Wp[(k/2)*N + n] = { f16(W[k][n]) , f16(W[k+1][n]) }  (lo = even K)
// Matches WMMA B layout: VGPR v<4, lanes 0-15 hold K={kk+2v, kk+2v+1} at
// N=lane (v>=4: K=kk+16+2(v-4), +1); lanes 16-31 hold K+8.
__global__ void k_pack_w(const float* __restrict__ W, uint32_t* __restrict__ Wp,
                         int K, int N) {
  int i = blockIdx.x * blockDim.x + threadIdx.x;
  int total = (K >> 1) * N;
  if (i >= total) return;
  int pr = i / N, n = i - pr * N;
  PackU pk;
  pk.h[0] = (_Float16)W[(2 * pr) * N + n];
  pk.h[1] = (_Float16)W[(2 * pr + 1) * N + n];
  Wp[i] = pk.u;
}

// ---------------- edge scatter: agg[dst] += feat[src]; deg[dst] += 1 -------
// one wave32 per edge; each lane moves 4 consecutive floats (b128 gather).
__global__ void k_aggregate(const float* __restrict__ feat,
                            const int* __restrict__ src,
                            const int* __restrict__ dst,
                            float* __restrict__ agg, float* deg, int E) {
  int wave = threadIdx.x >> 5;
  int lane = threadIdx.x & 31;
  int e = blockIdx.x * (blockDim.x >> 5) + wave;
  if (e >= E) return;
  int s = src[e], d = dst[e];
  const float4* row = (const float4*)(feat + (long)s * 128);
  float4 v = row[lane];
  float* outp = agg + (long)d * 128 + lane * 4;
  atomicAdd(outp + 0, v.x);
  atomicAdd(outp + 1, v.y);
  atomicAdd(outp + 2, v.z);
  atomicAdd(outp + 3, v.w);
  if (lane == 0 && deg) atomicAdd(deg + d, 1.0f);
}

// ---------------- mean: agg[r][k] /= max(deg[r],1) -------------------------
__global__ void k_divdeg(float* __restrict__ agg, const float* __restrict__ deg,
                         long total) {
  long i = (long)blockIdx.x * blockDim.x + threadIdx.x;
  if (i >= total) return;
  int r = (int)(i >> 7);
  float rd = 1.0f / fmaxf(deg[r], 1.0f);
  agg[i] *= rd;
}

// ---------------- fused SAGE layer GEMM via WMMA ---------------------------
// Out[r][c] = act( X[r,:] @ Ws[:,c] + G[r,:] @ Wn[:,c] + bias[c] )
// 16 rows per block; one wave per 16-col tile; K fully unrolled in 32-chunks.
// A tiles staged through LDS as packed-f16 dwords (v_cvt_pk_f16_f32).
__global__ void k_sage_gemm(const float* __restrict__ X,
                            const float* __restrict__ G,
                            const uint32_t* __restrict__ WpS,
                            const uint32_t* __restrict__ WpN,
                            const float* __restrict__ bias,
                            float* __restrict__ Out,
                            int nodes, int Ncols, int do_relu) {
  __shared__ uint32_t sX[16 * 16];   // 16 rows x 16 K-pair dwords (32 halfs)
  __shared__ uint32_t sG[16 * 16];
  const int tid  = threadIdx.x;
  const int lane = tid & 31;
  const int wave = tid >> 5;
  const int half = lane >> 4;       // lanes 16-31 handle K+8 / K+24 groups
  const int m    = lane & 15;
  const int r0   = blockIdx.x * 16;
  const int col  = wave * 16 + m;

  CFrag c;
#pragma unroll
  for (int j = 0; j < 8; ++j) c.f[j] = 0.0f;

#pragma unroll
  for (int kk = 0; kk < KDIM; kk += 32) {
    __syncthreads();
    // stage 16x32 f32 -> packed f16 pairs; dword i = row(i>>4) pair(i&15)
    for (int i = tid; i < 256; i += blockDim.x) {
      int row = i >> 4, kp = i & 15;
      int rr = r0 + row; if (rr >= nodes) rr = nodes - 1;   // clamp, keep EXEC full
      const float2 xv = *(const float2*)(X + (long)rr * KDIM + kk + 2 * kp);
      const float2 gv = *(const float2*)(G + (long)rr * KDIM + kk + 2 * kp);
      PackU px; px.h[0] = (_Float16)xv.x; px.h[1] = (_Float16)xv.y;
      PackU pg; pg.h[0] = (_Float16)gv.x; pg.h[1] = (_Float16)gv.y;
      sX[i] = px.u;
      sG[i] = pg.u;
    }
    __syncthreads();

    AFrag ax, ag, bs, bn;
#pragma unroll
    for (int v = 0; v < 8; ++v) {
      // A layout: VGPR v<4 -> K=2v(+8*half); v>=4 -> K=16+2(v-4)(+8*half)
      int pidx = (v < 4) ? (v + 4 * half) : (8 + (v - 4) + 4 * half);
      ax.u[v] = sX[m * 16 + pidx];
      ag.u[v] = sG[m * 16 + pidx];
      // B layout: pair row = kk/2 + pidx (same K order as A)
      int pr = (kk >> 1) + pidx;
      bs.u[v] = WpS[pr * Ncols + col];
      bn.u[v] = WpN[pr * Ncols + col];
    }
    c.v = __builtin_amdgcn_wmma_f32_16x16x32_f16(false, ax.v, false, bs.v,
                                                 (short)0, c.v, false, false);
    c.v = __builtin_amdgcn_wmma_f32_16x16x32_f16(false, ag.v, false, bn.v,
                                                 (short)0, c.v, false, false);
  }

  float bv = bias[col];
#pragma unroll
  for (int j = 0; j < 8; ++j) {
    int row = r0 + j + 8 * half;      // C/D layout: VGPR j -> M=j (+8 for half)
    if (row < nodes) {
      float val = c.f[j] + bv;
      if (do_relu) val = fmaxf(val, 0.0f);
      Out[(long)row * Ncols + col] = val;
    }
  }
}

// ---------------- edge dot over 64-dim rows; wave32 per edge ---------------
__global__ void k_edge_dot(const float* __restrict__ h2,
                           const int* __restrict__ src, const int* __restrict__ dst,
                           const int* __restrict__ nsrc, const int* __restrict__ ndst,
                           float* __restrict__ out, int E) {
  int wave = threadIdx.x >> 5;
  int lane = threadIdx.x & 31;
  int e = blockIdx.x * (blockDim.x >> 5) + wave;
  if (e >= 2 * E) return;
  int s, d;
  if (e < E) { s = src[e];      d = dst[e];      }
  else       { s = nsrc[e - E]; d = ndst[e - E]; }
  const float2* rs = (const float2*)(h2 + (long)s * 64);
  const float2* rd = (const float2*)(h2 + (long)d * 64);
  float2 a = rs[lane], b = rd[lane];
  float p = a.x * b.x + a.y * b.y;
#pragma unroll
  for (int off = 16; off > 0; off >>= 1) p += __shfl_xor(p, off, 32);
  if (lane == 0) out[e] = p;
}

// ---------------------------------------------------------------------------
extern "C" void kernel_launch(void* const* d_in, const int* in_sizes, int n_in,
                              void* d_out, int out_size, void* d_ws, size_t ws_size,
                              hipStream_t stream) {
  const float* feat = (const float*)d_in[0];
  const float* Ws1  = (const float*)d_in[1];
  const float* Wn1  = (const float*)d_in[2];
  const float* b1   = (const float*)d_in[3];
  const float* Ws2  = (const float*)d_in[4];
  const float* Wn2  = (const float*)d_in[5];
  const float* b2   = (const float*)d_in[6];
  const int*   src  = (const int*)d_in[7];
  const int*   dst  = (const int*)d_in[8];
  const int*   nsrc = (const int*)d_in[9];
  const int*   ndst = (const int*)d_in[10];
  float* out = (float*)d_out;

  const int nodes = in_sizes[0] / 128;
  const int E     = in_sizes[7];
  const int HID = 128, OUTD = 64;

  // workspace carve-out (256B aligned)
  char* ws = (char*)d_ws;
  size_t off = 0;
  auto take = [&](size_t bytes) -> void* {
    void* p = ws + off;
    off = (off + bytes + 255) & ~(size_t)255;
    return p;
  };
  uint32_t* Wp1s = (uint32_t*)take((size_t)64 * HID * 4);   // 128x128 -> 64x128 dwords
  uint32_t* Wp1n = (uint32_t*)take((size_t)64 * HID * 4);
  uint32_t* Wp2s = (uint32_t*)take((size_t)64 * OUTD * 4);  // 128x64 -> 64x64 dwords
  uint32_t* Wp2n = (uint32_t*)take((size_t)64 * OUTD * 4);
  float* deg = (float*)take((size_t)nodes * 4);
  float* agg = (float*)take((size_t)nodes * HID * 4);
  float* h   = (float*)take((size_t)nodes * HID * 4);
  float* h2  = (float*)take((size_t)nodes * OUTD * 4);

  // 1) pack weights to f16 K-pair layout
  {
    int t1 = (KDIM / 2) * HID;   // 8192
    int t2 = (KDIM / 2) * OUTD;  // 4096
    k_pack_w<<<(t1 + 255) / 256, 256, 0, stream>>>(Ws1, Wp1s, KDIM, HID);
    k_pack_w<<<(t1 + 255) / 256, 256, 0, stream>>>(Wn1, Wp1n, KDIM, HID);
    k_pack_w<<<(t2 + 255) / 256, 256, 0, stream>>>(Ws2, Wp2s, KDIM, OUTD);
    k_pack_w<<<(t2 + 255) / 256, 256, 0, stream>>>(Wn2, Wp2n, KDIM, OUTD);
  }

  const long aggN = (long)nodes * HID;
  const int edgeBlocks = (E + 7) / 8;
  const int gemmBlocks = (nodes + 15) / 16;

  // 2) layer-1 aggregation (also builds deg, reused for layer 2)
  k_zero<<<2048, 256, 0, stream>>>(agg, aggN);
  k_zero<<<256, 256, 0, stream>>>(deg, nodes);
  k_aggregate<<<edgeBlocks, 256, 0, stream>>>(feat, src, dst, agg, deg, E);
  k_divdeg<<<(int)((aggN + 255) / 256), 256, 0, stream>>>(agg, deg, aggN);

  // 3) layer-1 GEMM: h = relu(feat@Ws1 + agg@Wn1 + b1)   [nodes x 128]
  k_sage_gemm<<<gemmBlocks, 256, 0, stream>>>(feat, agg, Wp1s, Wp1n, b1,
                                              h, nodes, HID, 1);

  // 4) layer-2 aggregation of h (deg unchanged: same dst list)
  k_zero<<<2048, 256, 0, stream>>>(agg, aggN);
  k_aggregate<<<edgeBlocks, 256, 0, stream>>>(h, src, dst, agg, nullptr, E);
  k_divdeg<<<(int)((aggN + 255) / 256), 256, 0, stream>>>(agg, deg, aggN);

  // 5) layer-2 GEMM: h2 = h@Ws2 + agg@Wn2 + b2   [nodes x 64]
  k_sage_gemm<<<gemmBlocks, 128, 0, stream>>>(h, agg, Wp2s, Wp2n, b2,
                                              h2, nodes, OUTD, 0);

  // 6) positive + negative edge dots -> out[0..E) , out[E..2E)
  int dotBlocks = (2 * E + 7) / 8;
  k_edge_dot<<<dotBlocks, 256, 0, stream>>>(h2, src, dst, nsrc, ndst, out, E);
}
